// WeightOnlyLinear_81578608820986
// MI455X (gfx1250) — matrix operand order repaired and checked by reading, not compiled, not run on hardware
//
#include <hip/hip_runtime.h>
#include <stdint.h>

typedef __attribute__((ext_vector_type(16))) _Float16 v16h;
typedef __attribute__((ext_vector_type(8)))  _Float16 v8h;
typedef __attribute__((ext_vector_type(2)))  _Float16 v2h;
typedef __attribute__((ext_vector_type(8)))  float    v8f;

constexpr int IN     = 4096;
constexpr int OUT    = 11008;
constexpr int BATCH  = 32;
constexpr int GROUP  = 128;            // quant group along K
constexpr int KSTEP  = 32;             // f16 WMMA K
constexpr int NPK    = 8;              // int4 per int32
constexpr int WPR    = IN / NPK;       // 512 int32 per output row
constexpr int NGRP   = IN / GROUP;     // 32 scale groups
constexpr int SPLITK = 4;              // K-segments per tile (in-block, LDS-reduced)
constexpr int GPS    = NGRP / SPLITK;  // 8 groups per segment

// ---------------------------------------------------------------------------
// int4 -> f16 bit-trick dequant (exact).
// f16 pattern 0x6400|(q^8) == 1024+(q^8); pk_add(-1032) -> (q^8)-8 = sext int4.
// Per int32: 3 prep + 4x(v_perm, v_xor, v_pk_add_f16). No v_cvt anywhere.
// ---------------------------------------------------------------------------
union U32H2 { uint32_t u; v2h h; };

__device__ __forceinline__ v2h dq2(uint32_t hiB, uint32_t loB, uint32_t sel) {
  U32H2 t;
  // sel: dst byte0 <- lo byte r, byte2 <- hi byte r, bytes 1/3 <- 0x00.
  t.u = __builtin_amdgcn_perm(hiB, loB, sel) ^ 0x64086408u;
  const v2h offs = { (_Float16)(-1032.0f), (_Float16)(-1032.0f) };
  return t.h + offs;                    // v_pk_add_f16
}

__device__ __forceinline__ void dq_int32(uint32_t w, v2h* out4) {
  const uint32_t loB = w & 0x0F0F0F0Fu;          // nibbles 0,2,4,6 as bytes
  const uint32_t hiB = (w >> 4) & 0x0F0F0F0Fu;   // nibbles 1,3,5,7 as bytes
  out4[0] = dq2(hiB, loB, 0x0C040C00u);          // slots 0,1  <- n0,n1
  out4[1] = dq2(hiB, loB, 0x0C050C01u);          // slots 2,3  <- n2,n3
  out4[2] = dq2(hiB, loB, 0x0C060C02u);          // slots 4,5  <- n4,n5
  out4[3] = dq2(hiB, loB, 0x0C070C03u);          // slots 6,7  <- n6,n7
}

// ---------------------------------------------------------------------------
// Pre-pass: x fp32 -> f16 once (0.25 MB result, L2-resident, reused 688x).
// ---------------------------------------------------------------------------
__global__ __launch_bounds__(256) void cvt_x_f16_kernel(
    const float* __restrict__ x, _Float16* __restrict__ xh) {
  const int i = (blockIdx.x * blockDim.x + threadIdx.x) * 8;
  float4 f0 = *(const float4*)(x + i);
  float4 f1 = *(const float4*)(x + i + 4);
  v8h h;
  h[0] = (_Float16)f0.x; h[1] = (_Float16)f0.y;
  h[2] = (_Float16)f0.z; h[3] = (_Float16)f0.w;
  h[4] = (_Float16)f1.x; h[5] = (_Float16)f1.y;
  h[6] = (_Float16)f1.z; h[7] = (_Float16)f1.w;
  *(v8h*)(xh + i) = h;
}

// ---------------------------------------------------------------------------
// Fused kernel: block = 8 waves = 2 N-tiles x 4 K-segments.
// Each wave: one 16-column tile, one quarter of K (8 scale-groups), two 16x16
// f32 C tiles (M halves) sharing each B fragment. Group partials scaled in
// fp32 (reference-exact scale semantics). Split-K reduced through LDS in a
// fixed order (deterministic), bias added, single global store. No global
// partials, no second pass.
// ---------------------------------------------------------------------------
__global__ __launch_bounds__(256) void wql_int4_wmma_kernel(
    const _Float16* __restrict__ xh,      // [32, 4096] f16 (prestaged)
    const int*      __restrict__ pw,      // [11008, 512] packed int4
    const float*    __restrict__ scale,   // [11008, 32]
    const float*    __restrict__ bias,    // [11008]
    float*          __restrict__ out) {   // [32, 11008]
  // [tile-in-block*3 + (seg-1)][lane][4 x float4] : 12 KB of the 320 KB WGP LDS
  __shared__ float4 lred[6][32][4];

  const int lane = threadIdx.x & 31;
  const int wave = threadIdx.x >> 5;
  const int seg  = wave & 3;                // K segment 0..3
  const int tsub = wave >> 2;               // tile within block 0..1
  const int tile = blockIdx.x * 2 + tsub;   // 688 tiles = 344 blocks * 2
  const int n0   = tile * 16;
  const int nlo  = lane & 15;               // column within tile (B/C layout)
  const int col  = n0 + nlo;                // this lane's output column
  const int hi   = lane >> 4;               // lane half selects K sub-range

  v8f acc0 = {};   // rows 0..15
  v8f acc1 = {};   // rows 16..31

  const _Float16* arow0 = xh + (nlo)      * IN;   // A tile0: M = lane&15
  const _Float16* arow1 = xh + (16 + nlo) * IN;   // A tile1: M = 16+(lane&15)
  const int*      wrow  = pw + col * WPR;

  const int g0 = seg * GPS;
  for (int g = g0; g < g0 + GPS; ++g) {
    const int kg = g * GROUP;

    // Stream-prefetch next group's packed weights (used once -> NT).
    if (g + 1 < g0 + GPS)
      __builtin_prefetch((const void*)(wrow + (kg + GROUP) / NPK), 0, 0);

    v8f p0 = {};
    v8f p1 = {};

#pragma unroll
    for (int j = 0; j < GROUP / KSTEP; ++j) {
      const int k = kg + j * KSTEP;

      // ---- A fragments (16-bit A 16x32 layout): two b128 loads each ----
      // lane<16:  elems 0..15 = K k..k+15
      // lane>=16: elems 0..15 = K k+16..k+31
      const int abase = k + hi * 16;
      union { v16h v; v8h h[2]; } A0, A1;
      A0.h[0] = *(const v8h*)(arow0 + abase);
      A0.h[1] = *(const v8h*)(arow0 + abase + 8);
      A1.h[0] = *(const v8h*)(arow1 + abase);
      A1.h[1] = *(const v8h*)(arow1 + abase + 8);

      // ---- B fragment (16-bit B 32x16 layout) ----
      // lane<16: K k..k+15 of column col; lane>=16: K k+16..k+31.
      const int kb = k + hi * 16;
      const int2 w = *(const int2*)(wrow + kb / NPK);   // global_load_b64
      union { v16h v; v2h h2[8]; } B;
      dq_int32((uint32_t)w.x, &B.h2[0]);   // slots 0..7   (K kb..kb+7)
      dq_int32((uint32_t)w.y, &B.h2[4]);   // slots 8..15  (K kb+8..kb+15)

      p0 = __builtin_amdgcn_wmma_f32_16x16x32_f16(
          false, A0.v, false, B.v, (short)0, p0, false, false);
      p1 = __builtin_amdgcn_wmma_f32_16x16x32_f16(
          false, A1.v, false, B.v, (short)0, p1, false, false);
    }

    // Per-(column, group) scale in fp32; C layout keeps N = lane&15 fixed.
    const float s = scale[col * NGRP + g];
#pragma unroll
    for (int r = 0; r < 8; ++r) {
      acc0[r] += s * p0[r];
      acc1[r] += s * p1[r];
    }
  }

  // ---- In-block split-K reduction through LDS (fixed order, deterministic) --
  if (seg != 0) {
    const int widx = tsub * 3 + (seg - 1);
    lred[widx][lane][0] = make_float4(acc0[0], acc0[1], acc0[2], acc0[3]);
    lred[widx][lane][1] = make_float4(acc0[4], acc0[5], acc0[6], acc0[7]);
    lred[widx][lane][2] = make_float4(acc1[0], acc1[1], acc1[2], acc1[3]);
    lred[widx][lane][3] = make_float4(acc1[4], acc1[5], acc1[6], acc1[7]);
  }
  __syncthreads();
  if (seg == 0) {
#pragma unroll
    for (int w = 0; w < 3; ++w) {
      const int widx = tsub * 3 + w;
      float4 t0 = lred[widx][lane][0];
      float4 t1 = lred[widx][lane][1];
      float4 t2 = lred[widx][lane][2];
      float4 t3 = lred[widx][lane][3];
      acc0[0] += t0.x; acc0[1] += t0.y; acc0[2] += t0.z; acc0[3] += t0.w;
      acc0[4] += t1.x; acc0[5] += t1.y; acc0[6] += t1.z; acc0[7] += t1.w;
      acc1[0] += t2.x; acc1[1] += t2.y; acc1[2] += t2.z; acc1[3] += t2.w;
      acc1[4] += t3.x; acc1[5] += t3.y; acc1[6] += t3.z; acc1[7] += t3.w;
    }

    // Epilogue: bias + store. C/D layout: VGPR r holds row M = r + 8*hi.
    const float bv = bias[col];
#pragma unroll
    for (int r = 0; r < 8; ++r) {
      const int m = r + hi * 8;
      out[(size_t)m        * OUT + col] = acc0[r] + bv;   // rows 0..15
      out[(size_t)(16 + m) * OUT + col] = acc1[r] + bv;   // rows 16..31
    }
  }
}

// ---------------------------------------------------------------------------
// Launcher (graph-capture safe). Inputs: x, packed_weight, scale, bias.
// ws: 256 KB for x in f16 (only scratch needed now).
// ---------------------------------------------------------------------------
extern "C" void kernel_launch(void* const* d_in, const int* in_sizes, int n_in,
                              void* d_out, int out_size, void* d_ws, size_t ws_size,
                              hipStream_t stream) {
  const float* x     = (const float*)d_in[0];
  const int*   pw    = (const int*)d_in[1];
  const float* scale = (const float*)d_in[2];
  const float* bias  = (const float*)d_in[3];
  float*       out   = (float*)d_out;
  _Float16*    xh    = (_Float16*)d_ws;   // 32*4096*2 = 256 KB

  // 1) x fp32 -> f16 (131072 elems, 8 per thread -> 64 blocks)
  cvt_x_f16_kernel<<<(BATCH * IN) / (256 * 8), 256, 0, stream>>>(x, xh);

  // 2) fused dequant + WMMA GEMM + in-block split-K + bias:
  //    344 blocks x 8 waves = 2752 waves, zero global partial traffic.
  wql_int4_wmma_kernel<<<(OUT / 16) / 2, 256, 0, stream>>>(
      xh, pw, scale, bias, out);
}
